// Qwen3MergedAttention_7988639170813
// MI455X (gfx1250) — compile-verified, matrix operand order checked by reading, not run on hardware
//
#include <hip/hip_runtime.h>
#include <hip/hip_bf16.h>

// ---------------------------------------------------------------------------
// Qwen3 merged attention for MI455X (gfx1250).
// bf16 WMMA GEMMs + flash attention, both with double-buffered async-LDS
// staging (global_load_async_to_lds_b128 / s_wait_asynccnt).
// ---------------------------------------------------------------------------

typedef __bf16 bf16_t;
typedef __attribute__((ext_vector_type(16))) __bf16 v16bf;
typedef __attribute__((ext_vector_type(8)))  __bf16 v8bf;
typedef __attribute__((ext_vector_type(8)))  float  v8f;

#define HIDDEN 2048
#define NH 16
#define NKV 8
#define HD 128
#define SDEC 2048
#define SENC 2048
#define STOT 4096

__device__ __forceinline__ bf16_t f2bf(float f) {
  unsigned u = __builtin_bit_cast(unsigned, f);
  unsigned r = u + 0x7FFFu + ((u >> 16) & 1u);   // round-to-nearest-even
  unsigned short h = (unsigned short)(r >> 16);
  return __builtin_bit_cast(bf16_t, h);
}

__device__ __forceinline__ v8f vzero8() {
  v8f r;
#pragma unroll
  for (int i = 0; i < 8; i++) r[i] = 0.0f;
  return r;
}

__device__ __forceinline__ v16bf mk16(v8bf a, v8bf b) {
  v16bf r;
#pragma unroll
  for (int i = 0; i < 8; i++) { r[i] = a[i]; r[i + 8] = b[i]; }
  return r;
}

// 16-bit A/B WMMA fragment from row-major [rows x rstride] bf16 array.
// ISA 7.12.2 layout: lane holds row (lane&15); elements 0..7 = K d0..d0+7,
// elements 8..15 = K d0+16..d0+23, with d0 = chunk_base + ((lane>=16)?8:0).
__device__ __forceinline__ v16bf load_frag(const bf16_t* p, int row, int rstride, int d0) {
  const v8bf* q = (const v8bf*)(p + (size_t)row * rstride + d0);
  return mk16(q[0], q[2]);  // q[2] => +16 elements
}

__device__ __forceinline__ v8f wmma_bf16(v16bf a, v16bf b, v8f c) {
  return __builtin_amdgcn_wmma_f32_16x16x32_bf16(
      false, a, false, b, (short)0, c, false, false);
}

__device__ __forceinline__ float rowmax16(float x) {
#pragma unroll
  for (int m = 1; m < 16; m <<= 1) x = fmaxf(x, __shfl_xor(x, m, 32));
  return x;
}
__device__ __forceinline__ float rowsum16(float x) {
#pragma unroll
  for (int m = 1; m < 16; m <<= 1) x += __shfl_xor(x, m, 32);
  return x;
}

// CDNA5 async 16B global->LDS copy (per-lane), tracked by ASYNCcnt.
__device__ __forceinline__ void async_ld16(unsigned lds_off, const bf16_t* gptr) {
  asm volatile("global_load_async_to_lds_b128 %0, %1, off"
               :: "v"(lds_off), "v"((unsigned long long)(uintptr_t)gptr)
               : "memory");
}
__device__ __forceinline__ void wait_async0() {
  asm volatile("s_wait_asynccnt 0" ::: "memory");
}
__device__ __forceinline__ void wait_ds0() {
  asm volatile("s_wait_dscnt 0" ::: "memory");
}

// ---------------------------------------------------------------------------
// Prep kernels
// ---------------------------------------------------------------------------

__global__ void k_f32_to_bf16(const float* __restrict__ src, bf16_t* __restrict__ dst, int n) {
  int i = blockIdx.x * blockDim.x + threadIdx.x;
  if (i < n) dst[i] = f2bf(src[i]);
}

// W[K][N] (row-major) -> WT[N][K] bf16
__global__ void k_transpose_bf16(const float* __restrict__ W, bf16_t* __restrict__ WT,
                                 int K, int N) {
  int i = blockIdx.x * blockDim.x + threadIdx.x;
  if (i >= N * K) return;
  int n = i / K, k = i - n * K;
  WT[i] = f2bf(W[(size_t)k * N + n]);
}

// ---------------------------------------------------------------------------
// GEMM: C[M][N] (f32) = A[M][K] (bf16 row-major) * BT[N][K] (bf16 = B^T rows)
// Block = 128 threads (4 waves), 64x128 output tile. K stepped by 32 with
// double-buffered async-LDS staging: DMA of chunk i+1 overlaps WMMA on chunk i.
// ---------------------------------------------------------------------------
__global__ void k_gemm_bf16(const bf16_t* __restrict__ A, const bf16_t* __restrict__ BT,
                            float* __restrict__ C, int M, int N, int K) {
  __shared__ bf16_t As[2][64][32];    // 2 x 4 KB
  __shared__ bf16_t Bs[2][128][32];   // 2 x 8 KB

  int tid  = threadIdx.x;
  int wave = tid >> 5;
  int lane = tid & 31;
  int col  = lane & 15;
  int half = lane >> 4;
  int kb   = half * 8;
  int m0 = blockIdx.y * 64;
  int n0 = blockIdx.x * 128;

  auto fill = [&](int k0, int buf) {
    unsigned ab = (unsigned)(uintptr_t)&As[buf][0][0];
    unsigned bb = (unsigned)(uintptr_t)&Bs[buf][0][0];
#pragma unroll
    for (int t = 0; t < 2; t++) {                 // A tile: 64x32 = 256 x16B
      int j = tid + t * 128;
      int r = j >> 2, c8 = (j & 3) * 8;
      async_ld16(ab + (unsigned)((r * 32 + c8) * 2),
                 A + (size_t)(m0 + r) * K + k0 + c8);
    }
#pragma unroll
    for (int t = 0; t < 4; t++) {                 // B tile: 128x32 = 512 x16B
      int j = tid + t * 128;
      int r = j >> 2, c8 = (j & 3) * 8;
      async_ld16(bb + (unsigned)((r * 32 + c8) * 2),
                 BT + (size_t)(n0 + r) * K + k0 + c8);
    }
  };

  v8f acc[8];
#pragma unroll
  for (int t = 0; t < 8; t++) acc[t] = vzero8();

  int nk = K >> 5;
  fill(0, 0);
  wait_async0();
  __syncthreads();

  for (int ki = 0; ki < nk; ki++) {
    int cur = ki & 1;
    if (ki + 1 < nk) fill((ki + 1) * 32, cur ^ 1);   // prefetch overlaps WMMA

    v16bf a = load_frag(&As[cur][0][0], wave * 16 + col, 32, kb);
#pragma unroll
    for (int t = 0; t < 8; t++) {
      v16bf b = load_frag(&Bs[cur][0][0], t * 16 + col, 32, kb);
      acc[t] = wmma_bf16(a, b, acc[t]);
    }
    wait_async0();
    __syncthreads();
  }

#pragma unroll
  for (int t = 0; t < 8; t++)
#pragma unroll
    for (int v = 0; v < 8; v++)
      C[(size_t)(m0 + wave * 16 + v + half * 8) * N + (n0 + t * 16 + col)] = acc[t][v];
}

// ---------------------------------------------------------------------------
// Fused per-head RMSNorm + RoPE, pack head-major bf16 [heads][Sstride][128].
// One wave per (s, head) row; lane owns rotate-half pairs (p, p+64).
// ---------------------------------------------------------------------------
__global__ void k_norm_rope_pack(const float* __restrict__ src, const float* __restrict__ w,
                                 bf16_t* __restrict__ dst, int S, int heads,
                                 int dstSstride, int dstSoffset) {
  int wid  = blockIdx.x * (blockDim.x >> 5) + (threadIdx.x >> 5);
  int lane = threadIdx.x & 31;
  if (wid >= S * heads) return;
  int s = wid / heads, h = wid - s * heads;

  const float* row = src + (size_t)s * heads * HD + h * HD;
  int p0 = lane * 2;
  float x0 = row[p0], x1 = row[p0 + 1], x2 = row[p0 + 64], x3 = row[p0 + 65];
  float ss = x0 * x0 + x1 * x1 + x2 * x2 + x3 * x3;
#pragma unroll
  for (int m = 1; m < 32; m <<= 1) ss += __shfl_xor(ss, m, 32);
  float inv = rsqrtf(ss * (1.0f / 128.0f) + 1e-6f);

  float n0 = x0 * inv * w[p0],      n1 = x1 * inv * w[p0 + 1];
  float n2 = x2 * inv * w[p0 + 64], n3 = x3 * inv * w[p0 + 65];

  float pos = (float)s;
  float f0 = pos * powf(1000000.0f, -(float)(2 * p0) / 128.0f);
  float f1 = pos * powf(1000000.0f, -(float)(2 * (p0 + 1)) / 128.0f);
  float c0 = cosf(f0), s0 = sinf(f0), c1 = cosf(f1), s1 = sinf(f1);

  bf16_t* drow = dst + ((size_t)h * dstSstride + dstSoffset + s) * HD;
  drow[p0]      = f2bf(n0 * c0 - n2 * s0);
  drow[p0 + 64] = f2bf(n2 * c0 + n0 * s0);
  drow[p0 + 1]  = f2bf(n1 * c1 - n3 * s1);
  drow[p0 + 65] = f2bf(n3 * c1 + n1 * s1);
}

// V pack: Vt[kv][d][s] = {Vd|Ve}[s][kv*128+d] bf16 (transposed for PV B-frags)
__global__ void k_pack_v(const float* __restrict__ Vd, const float* __restrict__ Ve,
                         bf16_t* __restrict__ Vt) {
  size_t i = (size_t)blockIdx.x * blockDim.x + threadIdx.x;
  if (i >= (size_t)NKV * HD * STOT) return;
  int s  = (int)(i & (STOT - 1));
  int d  = (int)((i >> 12) & (HD - 1));
  int kv = (int)(i >> 19);
  float v = (s < SDEC) ? Vd[(size_t)s * (NKV * HD) + kv * HD + d]
                       : Ve[(size_t)(s - SDEC) * (NKV * HD) + kv * HD + d];
  Vt[i] = f2bf(v);
}

// ---------------------------------------------------------------------------
// Flash attention. Block = 4 waves (128 threads) covering 64 q rows of head h.
// 64-key K/V chunks double-buffered in LDS via async copies; softmax update
// amortized per 64 keys. Grid = (SDEC/64, NH).
// Qc [H][2048][128], Kc [KVH][4096][128], Vt [KVH][128][4096], all bf16.
// ---------------------------------------------------------------------------
__global__ void k_attention(const bf16_t* __restrict__ Qc, const bf16_t* __restrict__ Kc,
                            const bf16_t* __restrict__ Vt,
                            const int* __restrict__ dmask, const int* __restrict__ emask,
                            bf16_t* __restrict__ attnC) {
  __shared__ bf16_t Ks[2][64][128];   // 2 x 16 KB: K chunk, key-major
  __shared__ bf16_t Vs[2][128][64];   // 2 x 16 KB: V^T chunk, [d][key]
  __shared__ bf16_t Ps[4][16][64];    //     8 KB: per-wave probabilities

  int tid  = threadIdx.x;
  int wave = tid >> 5;
  int lane = tid & 31;
  int col  = lane & 15;
  int half = lane >> 4;
  int kb   = half * 8;

  int h = blockIdx.y, kvh = h >> 1;
  int qbase = blockIdx.x * 64 + wave * 16;

  const bf16_t* Qh = Qc + (size_t)h   * SDEC * HD;
  const bf16_t* Kh = Kc + (size_t)kvh * STOT * HD;
  const bf16_t* Vh = Vt + (size_t)kvh * HD * STOT;
  bf16_t* Psw = &Ps[wave][0][0];

  auto fillKV = [&](int chunk, int buf) {
    int kbase = chunk * 64;
    unsigned ksb = (unsigned)(uintptr_t)&Ks[buf][0][0];
    unsigned vsb = (unsigned)(uintptr_t)&Vs[buf][0][0];
#pragma unroll
    for (int t = 0; t < 8; t++) {               // K chunk: 64x128 = 1024 x16B
      int j = tid + t * 128;
      int r = j >> 4, dc = (j & 15) * 8;
      async_ld16(ksb + (unsigned)((r * HD + dc) * 2),
                 Kh + (size_t)(kbase + r) * HD + dc);
    }
#pragma unroll
    for (int t = 0; t < 8; t++) {               // V^T chunk: 128x64 = 1024 x16B
      int j = tid + t * 128;
      int d = j >> 3, kk = (j & 7) * 8;
      async_ld16(vsb + (unsigned)((d * 64 + kk) * 2),
                 Vh + (size_t)d * STOT + kbase + kk);
    }
  };

  v16bf aq[4];
#pragma unroll
  for (int kc = 0; kc < 4; kc++) aq[kc] = load_frag(Qh, qbase + col, HD, kc * 32 + kb);

  float mrow[8], lrow[8];
  v8f acc[8];
#pragma unroll
  for (int v = 0; v < 8; v++) { mrow[v] = -1e30f; lrow[v] = 0.0f; }
#pragma unroll
  for (int t = 0; t < 8; t++) acc[t] = vzero8();

  const float scale = 0.08838834764831845f;  // 1/sqrt(128)
  int dec_chunks = blockIdx.x + 1;           // causal: chunks of 64 keys
  int total = dec_chunks + SENC / 64;        // + 32 encoder chunks
  auto chunk_of = [&](int i) { return (i < dec_chunks) ? i : (SDEC / 64 + i - dec_chunks); };

  fillKV(chunk_of(0), 0);
  wait_async0();
  __syncthreads();

  for (int i = 0; i < total; i++) {            // block-uniform bounds
    int cur = i & 1;
    if (i + 1 < total) fillKV(chunk_of(i + 1), cur ^ 1);  // overlap DMA w/ compute
    int kbase = chunk_of(i) * 64;

    // --- scores: 4 tiles x 4 k-chunks = 16 WMMA ---------------------------
    v8f sv[4];
#pragma unroll
    for (int st = 0; st < 4; st++) {
      sv[st] = vzero8();
#pragma unroll
      for (int kc = 0; kc < 4; kc++) {
        v16bf bk = load_frag(&Ks[cur][0][0], st * 16 + col, HD, kc * 32 + kb);
        sv[st] = wmma_bf16(aq[kc], bk, sv[st]);
      }
    }

    bool kvalid[4];
#pragma unroll
    for (int st = 0; st < 4; st++) {
      int key = kbase + st * 16 + col;
      kvalid[st] = (key < SDEC) ? (dmask[key] != 0) : (emask[key - SDEC] != 0);
    }

    // --- one online-softmax pass per 64 keys ------------------------------
#pragma unroll
    for (int v = 0; v < 8; v++) {
      int qrow = qbase + v + half * 8;
      float sl[4];
      float lm = -1e30f;
#pragma unroll
      for (int st = 0; st < 4; st++) {
        int key = kbase + st * 16 + col;
        float s = sv[st][v] * scale;
        bool ok = kvalid[st] && (key >= SDEC || key <= qrow);
        sl[st] = ok ? s : -1e30f;
        lm = fmaxf(lm, sl[st]);
      }
      lm = rowmax16(lm);
      float nm = fmaxf(mrow[v], lm);
      float alpha = __expf(mrow[v] - nm);
      float ps = 0.0f;
#pragma unroll
      for (int st = 0; st < 4; st++) {
        float p = __expf(sl[st] - nm);
        ps += p;
        Psw[(v + half * 8) * 64 + st * 16 + col] = f2bf(p);
      }
      lrow[v] = lrow[v] * alpha + rowsum16(ps);
      mrow[v] = nm;
#pragma unroll
      for (int t = 0; t < 8; t++) acc[t][v] *= alpha;
    }
    wait_ds0();

    // --- PV: 2 k-subchunks x 8 n-tiles = 16 WMMA --------------------------
#pragma unroll
    for (int kk = 0; kk < 2; kk++) {
      v16bf ap = load_frag(Psw, col, 64, kk * 32 + kb);
#pragma unroll
      for (int t = 0; t < 8; t++) {
        v16bf bv = load_frag(&Vs[cur][0][0], t * 16 + col, 64, kk * 32 + kb);
        acc[t] = wmma_bf16(ap, bv, acc[t]);
      }
    }
    wait_async0();       // prefetch of next chunk finished
    __syncthreads();     // all waves done reading cur before it is refilled
  }

#pragma unroll
  for (int t = 0; t < 8; t++)
#pragma unroll
    for (int v = 0; v < 8; v++) {
      int qrow = qbase + v + half * 8;
      float o = acc[t][v] / lrow[v];
      attnC[(size_t)qrow * (NH * HD) + h * HD + t * 16 + col] = f2bf(o);
    }
}

// ---------------------------------------------------------------------------
// Host launcher
// ---------------------------------------------------------------------------
extern "C" void kernel_launch(void* const* d_in, const int* in_sizes, int n_in,
                              void* d_out, int out_size, void* d_ws, size_t ws_size,
                              hipStream_t stream) {
  const float* dec   = (const float*)d_in[0];
  const float* enc   = (const float*)d_in[1];
  const float* Wq    = (const float*)d_in[2];
  const float* Wk    = (const float*)d_in[3];
  const float* Wv    = (const float*)d_in[4];
  const float* Wo    = (const float*)d_in[5];
  const float* qnw   = (const float*)d_in[6];
  const float* knw   = (const float*)d_in[7];
  const int*   dmask = (const int*)d_in[8];
  const int*   emask = (const int*)d_in[9];
  float* out = (float*)d_out;

  char* ws = (char*)d_ws;
  size_t off = 0;
  auto alloc = [&](size_t bytes) -> char* {
    char* p = ws + off;
    off += (bytes + 255) & ~(size_t)255;
    return p;
  };

  bf16_t* decb  = (bf16_t*)alloc((size_t)SDEC * HIDDEN * 2);
  bf16_t* encb  = (bf16_t*)alloc((size_t)SENC * HIDDEN * 2);
  bf16_t* WqT   = (bf16_t*)alloc((size_t)HIDDEN * HIDDEN * 2);
  bf16_t* WkT   = (bf16_t*)alloc((size_t)(NKV * HD) * HIDDEN * 2);
  bf16_t* WvT   = (bf16_t*)alloc((size_t)(NKV * HD) * HIDDEN * 2);
  bf16_t* WoT   = (bf16_t*)alloc((size_t)HIDDEN * (NH * HD) * 2);
  float*  Qf    = (float*) alloc((size_t)SDEC * NH * HD * 4);
  float*  Kdf   = (float*) alloc((size_t)SDEC * NKV * HD * 4);
  float*  Vdf   = (float*) alloc((size_t)SDEC * NKV * HD * 4);
  float*  Kef   = (float*) alloc((size_t)SENC * NKV * HD * 4);
  float*  Vef   = (float*) alloc((size_t)SENC * NKV * HD * 4);
  bf16_t* Qc    = (bf16_t*)alloc((size_t)NH * SDEC * HD * 2);
  bf16_t* Kc    = (bf16_t*)alloc((size_t)NKV * STOT * HD * 2);
  bf16_t* Vt    = (bf16_t*)alloc((size_t)NKV * HD * STOT * 2);
  bf16_t* attnC = (bf16_t*)alloc((size_t)SDEC * NH * HD * 2);
  (void)ws_size; (void)n_in; (void)in_sizes; (void)out_size;

  // 1) fp32 -> bf16 hidden states
  {
    int n = SDEC * HIDDEN;
    k_f32_to_bf16<<<(n + 255) / 256, 256, 0, stream>>>(dec, decb, n);
    k_f32_to_bf16<<<(n + 255) / 256, 256, 0, stream>>>(enc, encb, n);
  }
  // 2) weight transpose + convert
  {
    int n = HIDDEN * HIDDEN;
    k_transpose_bf16<<<(n + 255) / 256, 256, 0, stream>>>(Wq, WqT, HIDDEN, HIDDEN);
    k_transpose_bf16<<<(n + 255) / 256, 256, 0, stream>>>(Wo, WoT, NH * HD, HIDDEN);
    int m = HIDDEN * NKV * HD;
    k_transpose_bf16<<<(m + 255) / 256, 256, 0, stream>>>(Wk, WkT, HIDDEN, NKV * HD);
    k_transpose_bf16<<<(m + 255) / 256, 256, 0, stream>>>(Wv, WvT, HIDDEN, NKV * HD);
  }
  // 3) projection GEMMs (bf16 WMMA, async-LDS double buffered)
  {
    dim3 blk(128);
    k_gemm_bf16<<<dim3((NH * HD) / 128, SDEC / 64), blk, 0, stream>>>(decb, WqT, Qf,  SDEC, NH * HD,  HIDDEN);
    k_gemm_bf16<<<dim3((NKV * HD) / 128, SDEC / 64), blk, 0, stream>>>(decb, WkT, Kdf, SDEC, NKV * HD, HIDDEN);
    k_gemm_bf16<<<dim3((NKV * HD) / 128, SDEC / 64), blk, 0, stream>>>(decb, WvT, Vdf, SDEC, NKV * HD, HIDDEN);
    k_gemm_bf16<<<dim3((NKV * HD) / 128, SENC / 64), blk, 0, stream>>>(encb, WkT, Kef, SENC, NKV * HD, HIDDEN);
    k_gemm_bf16<<<dim3((NKV * HD) / 128, SENC / 64), blk, 0, stream>>>(encb, WvT, Vef, SENC, NKV * HD, HIDDEN);
  }
  // 4) RMSNorm + RoPE, pack head-major bf16
  {
    int wavesQ = SDEC * NH;
    k_norm_rope_pack<<<(wavesQ + 7) / 8, 256, 0, stream>>>(Qf, qnw, Qc, SDEC, NH, SDEC, 0);
    int wavesK = SDEC * NKV;
    k_norm_rope_pack<<<(wavesK + 7) / 8, 256, 0, stream>>>(Kdf, knw, Kc, SDEC, NKV, STOT, 0);
    k_norm_rope_pack<<<(wavesK + 7) / 8, 256, 0, stream>>>(Kef, knw, Kc, SENC, NKV, STOT, SDEC);
  }
  // 5) V pack (transpose + bf16)
  {
    size_t n = (size_t)NKV * HD * STOT;
    k_pack_v<<<(unsigned)((n + 255) / 256), 256, 0, stream>>>(Vdf, Vef, Vt);
  }
  // 6) flash attention: 4 waves/block, double-buffered async-LDS K/V
  k_attention<<<dim3(SDEC / 64, NH), 128, 0, stream>>>(Qc, Kc, Vt, dmask, emask, attnC);

  // 7) output projection -> f32 d_out
  k_gemm_bf16<<<dim3(HIDDEN / 128, SDEC / 64), 128, 0, stream>>>(attnC, WoT, out, SDEC, HIDDEN, NH * HD);
}
